// TripletAttention_21809843929813
// MI455X (gfx1250) — matrix operand order repaired and checked by reading, not compiled, not run on hardware
//
#include <hip/hip_runtime.h>
#include <hip/hip_bf16.h>

// ---------------------------------------------------------------------------
// TripletAttention for MI455X (gfx1250, wave32, WMMA bf16 16x16x32 f32-accum)
//
// Roofline: dominant term is B*H*N^3*(D+1)*2 ~= 58 GFLOP of dense matmul; all
// operands (~12 MB) are L2-resident (192 MB L2, 23.3 TB/s HBM) -> matrix-core
// bound, all heavy math via v_wmma_f32_16x16x32_bf16.
// k_triplet v3: 5 d-channels per workgroup amortize the eEFt panel (raw panel
// LDS-resident per column pass; per-channel v1 scaling applied at fragment
// time, co-executing under the XDL WMMAs). Global B traffic drops 5x.
// ---------------------------------------------------------------------------

#define NTOK 384
#define CDIM 256
#define HN   4
#define DDIM 64
#define BB   2
#define BH   (BB*HN)

// LDS padding strides (elements) for bank-conflict-free b128 access:
// 392*2B = 784B = 196 banks ~ 4 mod 64 -> 16 consecutive rows cover banks 0..63
#define ASTRIDE 392
#define GSTRIDE 392

typedef __bf16 bf16_t;
typedef __attribute__((ext_vector_type(16))) __bf16 v16bf;
typedef __attribute__((ext_vector_type(8)))  float  v8f;

__device__ __forceinline__ v8f v8f_zero() {
  v8f z;
#pragma unroll
  for (int e = 0; e < 8; e++) z[e] = 0.f;
  return z;
}

// 16 bf16 (32B) load as two 128-bit loads (works for LDS and global pointers)
__device__ __forceinline__ v16bf ld16(const bf16_t* p) {
  union { int4 q[2]; v16bf v; } u;
  u.q[0] = *reinterpret_cast<const int4*>(p);
  u.q[1] = *reinterpret_cast<const int4*>(p + 8);
  return u.v;
}

// ---------------------------------------------------------------------------
// K0: fp32 -> bf16 convert of x and the 5 needed W row-chunks (a,b,c,v1,v2)
// ---------------------------------------------------------------------------
__global__ void k_convert(const float* __restrict__ x, const float* __restrict__ W,
                          bf16_t* __restrict__ xbf, bf16_t* __restrict__ Wbf) {
  const int NX = BB * NTOK * CDIM;           // 196608
  int idx = blockIdx.x * 256 + threadIdx.x;
  if (idx < NX) xbf[idx] = (bf16_t)x[idx];
  int i2 = idx - NX;
  if (i2 >= 0 && i2 < 1280 * CDIM) {
    int rr = i2 >> 8, cc = i2 & 255;
    int chunk = rr >> 8;                      // 0..4
    int srcChunk = (chunk < 3) ? chunk : (chunk + 3);  // a,b,c,v1,v2 -> 0,1,2,6,7
    Wbf[i2] = (bf16_t)W[(size_t)(srcChunk * 256 + (rr & 255)) * CDIM + cc];
  }
}

// ---------------------------------------------------------------------------
// K1: projection GEMM  Y[768 x 1280] = xbf[768 x 256] @ Wbf^T, scatter outputs
//   chunks 0..2 -> a/b/c  bf16 [bh][N][D]
//   chunk  3    -> v1t    bf16 [bh][D][N]   (transposed for per-d row access)
//   chunk  4    -> v2t    f32  [bh][D][N]
// ---------------------------------------------------------------------------
__global__ void __launch_bounds__(256) k_proj(const bf16_t* __restrict__ xbf,
                                              const bf16_t* __restrict__ Wbf,
                                              bf16_t* __restrict__ abc,
                                              bf16_t* __restrict__ v1t,
                                              float*  __restrict__ v2t) {
  __shared__ __align__(16) bf16_t sA[128 * 32];
  __shared__ __align__(16) bf16_t sB[128 * 32];
  const int m0 = blockIdx.x * 128;
  const int i0 = blockIdx.y * 128;
  const int t = threadIdx.x;
  const int wave = t >> 5, lane = t & 31;
  const int r = t >> 1, half = t & 1;

  v8f acc[8];
#pragma unroll
  for (int q = 0; q < 8; q++) acc[q] = v8f_zero();

  for (int ks = 0; ks < 8; ks++) {
    const int kb = ks * 32;
    { // stage A (x tile), K-contiguous per row
      const bf16_t* src = xbf + (size_t)(i0 + r) * CDIM + kb + half * 16;
      int4 q0 = *(const int4*)src;
      int4 q1 = *(const int4*)(src + 8);
      int4* dst = (int4*)&sA[r * 32];
      dst[half * 2]     = q0;
      dst[half * 2 + 1] = q1;
    }
    { // stage B (W tile), K-contiguous per column
      const bf16_t* src = Wbf + (size_t)(m0 + r) * CDIM + kb + half * 16;
      int4* dst = (int4*)&sB[r * 32 + half * 16];
      dst[0] = *(const int4*)src;
      dst[1] = *(const int4*)(src + 8);
    }
    __syncthreads();
    v16bf af = ld16(&sA[(wave * 16 + (lane & 15)) * 32 + ((lane >> 4) * 16)]);
#pragma unroll
    for (int ct = 0; ct < 8; ct++) {
      v16bf bv = ld16(&sB[(ct * 16 + (lane & 15)) * 32 + ((lane >> 4) * 16)]);
      acc[ct] = __builtin_amdgcn_wmma_f32_16x16x32_bf16(false, af, false, bv,
                                                        (short)0, acc[ct], false, false);
    }
    __syncthreads();
  }

  const int rowBase = i0 + wave * 16 + ((lane >> 4) << 3);
#pragma unroll
  for (int ct = 0; ct < 8; ct++) {
    const int m = m0 + ct * 16 + (lane & 15);
    const int chunk = m >> 8, h = (m >> 6) & 3, dd = m & 63;
#pragma unroll
    for (int g = 0; g < 8; g++) {
      const int row = rowBase + g;
      const int b = (row >= NTOK) ? 1 : 0;
      const int nn = row - b * NTOK;
      const int bh = b * HN + h;
      const float v = acc[ct][g];
      if (chunk < 3)
        abc[((size_t)chunk * BH + bh) * (NTOK * DDIM) + (size_t)nn * DDIM + dd] = (bf16_t)v;
      else if (chunk == 3)
        v1t[((size_t)bh * DDIM + dd) * NTOK + nn] = (bf16_t)v;
      else
        v2t[((size_t)bh * DDIM + dd) * NTOK + nn] = v;
    }
  }
}

// ---------------------------------------------------------------------------
// K2: logits. Per bh, three 384x384 GEMMs (K=64) + exp:
//   which==0: eAB[i,j]  = exp(s * a_i . b_j)  -> bf16
//   which==1: eCD[i,k]  = exp(s * b_i . c_k)  -> f32
//   which==2: eEFt[k,j] = exp(s * a_k . c_j)  -> bf16   (eEF stored transposed)
// ---------------------------------------------------------------------------
__global__ void __launch_bounds__(256) k_logits(const bf16_t* __restrict__ abc,
                                                bf16_t* __restrict__ eAB,
                                                float*  __restrict__ eCD,
                                                bf16_t* __restrict__ eEFt) {
  const int which = blockIdx.x;               // 0,1,2
  const int ti = blockIdx.y / 3, tj = blockIdx.y % 3;
  const int bh = blockIdx.z;
  const int wave = threadIdx.x >> 5, lane = threadIdx.x & 31;
  const size_t mhd = (size_t)NTOK * DDIM;
  const bf16_t* Abase = abc + ((which == 1) ? (size_t)BH * mhd : 0) + (size_t)bh * mhd;
  const bf16_t* Bbase = abc + ((which == 0) ? (size_t)BH * mhd : 2 * (size_t)BH * mhd) + (size_t)bh * mhd;
  const int i0 = ti * 128 + wave * 16;
  const int cb = tj * 128;

  v8f acc[8];
#pragma unroll
  for (int q = 0; q < 8; q++) acc[q] = v8f_zero();

#pragma unroll
  for (int ks = 0; ks < 2; ks++) {
    const int kb = ks * 32;
    // A fragment: low lanes K{0-15}, high lanes K{16-31}
    const bf16_t* ap = Abase + (size_t)(i0 + (lane & 15)) * DDIM + kb + ((lane >> 4) * 16);
    union { int4 q[2]; v16bf v; } ua;
    ua.q[0] = *(const int4*)ap;
    ua.q[1] = *(const int4*)(ap + 8);
#pragma unroll
    for (int ct = 0; ct < 8; ct++) {
      const bf16_t* bp = Bbase + (size_t)(cb + ct * 16 + (lane & 15)) * DDIM + kb + ((lane >> 4) * 16);
      v16bf bv = ld16(bp);
      acc[ct] = __builtin_amdgcn_wmma_f32_16x16x32_bf16(false, ua.v, false, bv,
                                                        (short)0, acc[ct], false, false);
    }
  }

#pragma unroll
  for (int ct = 0; ct < 8; ct++) {
    const int col = cb + ct * 16 + (lane & 15);
#pragma unroll
    for (int g = 0; g < 8; g++) {
      const int row = i0 + ((lane >> 4) << 3) + g;
      const float e = __expf(acc[ct][g] * 0.125f);   // scale = D^-0.5 = 1/8
      const size_t o = (size_t)bh * NTOK * NTOK + (size_t)row * NTOK + col;
      if (which == 0)      eAB[o]  = (bf16_t)e;
      else if (which == 1) eCD[o]  = e;
      else                 eEFt[o] = (bf16_t)e;
    }
  }
}

// ---------------------------------------------------------------------------
// K3: fused heavy phase. Workgroup = (chan-group of 5, rowblock, bh).
// For each chan: P = eAB @ (diag over j of v1[:,chan]) @ eEF;
//                result[i] = sum_k P[i,k] * v2[k,chan] * eCD[i,k]
// chan 64 (in group 12) = denominator: v1=v2=1.
// LDS: sA = 128x384 eAB panel (j-contiguous, no swizzle, staged once, 98KB)
//      sG = 128x384 raw eEFt k-panel per column pass (98KB), shared by all
//           5 channels; v1 scaling applied per fragment in VALU (pk-muls
//           co-execute under the XDL WMMA chain).
// Barriers: 1 in the first column pass, 2 in each later one (5 total). ~196KB LDS -> 1 WG per 320KB WGP.
// ---------------------------------------------------------------------------
__global__ void __launch_bounds__(256) k_triplet(const bf16_t* __restrict__ eAB,
                                                 const bf16_t* __restrict__ eEFt,
                                                 const float*  __restrict__ eCD,
                                                 const bf16_t* __restrict__ v1t,
                                                 const float*  __restrict__ v2t,
                                                 float* __restrict__ outp,
                                                 float* __restrict__ denp) {
  __shared__ __align__(16) bf16_t sA[128 * ASTRIDE];
  __shared__ __align__(16) bf16_t sG[128 * GSTRIDE];
  const int gx = blockIdx.x;                   // chan group: chans gx*5 .. gx*5+4
  const int i0 = blockIdx.y * 128;
  const int bh = blockIdx.z;
  const int b = bh >> 2, h = bh & 3;
  const int t = threadIdx.x;
  const int wave = t >> 5, lane = t & 31;
  const int r = t >> 1, half = t & 1;

  const bf16_t* Aab = eAB  + (size_t)bh * NTOK * NTOK;   // [i][j]
  const bf16_t* Gt  = eEFt + (size_t)bh * NTOK * NTOK;   // [k][j]
  const float*  Ecd = eCD  + (size_t)bh * NTOK * NTOK;   // [i][k]

  // ---- stage entire A panel once (j-contiguous per row)
  {
    const bf16_t* srcRow = Aab + (size_t)(i0 + r) * NTOK + half * 16;
    int4* dstRow = (int4*)&sA[r * ASTRIDE];    // ASTRIDE/8 = 49 int4 per row
#pragma unroll
    for (int js = 0; js < 12; js++) {
      int4 q0 = *(const int4*)(srcRow + js * 32);
      int4 q1 = *(const int4*)(srcRow + js * 32 + 8);
      int4* dst = dstRow + js * 4;
      dst[half * 2]     = q0;
      dst[half * 2 + 1] = q1;
    }
  }

  float numacc[5][8];
#pragma unroll
  for (int c = 0; c < 5; c++)
#pragma unroll
    for (int g = 0; g < 8; g++) numacc[c][g] = 0.f;

  for (int cp = 0; cp < 3; cp++) {             // column passes over k
    const int kbase = cp * 128;
    if (cp > 0) __syncthreads();               // sG reads of previous pass done
    { // stage raw eEFt k-panel [128 k][384 j] (no swizzle: B frags are j-contig)
      const bf16_t* srcRow = Gt + (size_t)(kbase + r) * NTOK + half * 16;
      bf16_t* dstRow = &sG[r * GSTRIDE + half * 16];
#pragma unroll
      for (int js = 0; js < 12; js++) {
        int4 q0 = *(const int4*)(srcRow + js * 32);
        int4 q1 = *(const int4*)(srcRow + js * 32 + 8);
        int4* dst = (int4*)(dstRow + js * 32);
        dst[0] = q0;
        dst[1] = q1;
      }
    }
    __syncthreads();
    // pull the eCD epilogue region for this pass toward the WGP caches
    __builtin_prefetch(&Ecd[(size_t)(i0 + wave * 16 + (lane & 15)) * NTOK + kbase], 0, 0);

#pragma unroll
    for (int c = 0; c < 5; c++) {              // 5 d-channels share the panels
      const int chan = gx * 5 + c;             // 0..64; 64 = denominator
      const bool scaled = (chan < DDIM);
      const bf16_t* v1row = v1t + ((size_t)bh * DDIM + (scaled ? chan : 0)) * NTOK;
      const float*  v2row = v2t + ((size_t)bh * DDIM + (scaled ? chan : 0)) * NTOK;

      v8f acc[8];
#pragma unroll
      for (int q = 0; q < 8; q++) acc[q] = v8f_zero();

      for (int js = 0; js < 12; js++) {        // contraction over j
        const int jb = js * 32;
        v16bf af = ld16(&sA[(wave * 16 + (lane & 15)) * ASTRIDE + jb + ((lane >> 4) * 16)]);
        v16bf vs;
        if (scaled) vs = ld16(v1row + jb + ((lane >> 4) * 16));  // v1[j] slice, j-contig
#pragma unroll
        for (int ct = 0; ct < 8; ct++) {
          v16bf bv = ld16(&sG[(ct * 16 + (lane & 15)) * GSTRIDE + jb + ((lane >> 4) * 16)]);
          if (scaled) {
#pragma unroll
            for (int e = 0; e < 16; e++)       // lowers to packed bf16 muls;
              bv[e] = (bf16_t)((float)bv[e] * (float)vs[e]);  // co-executes w/ WMMA
          }
          acc[ct] = __builtin_amdgcn_wmma_f32_16x16x32_bf16(false, af, false, bv,
                                                            (short)0, acc[ct], false, false);
        }
      }

      // fused epilogue: weighted reduce over this pass's k range
      const int rowLocal = i0 + wave * 16 + ((lane >> 4) << 3);
#pragma unroll
      for (int ct = 0; ct < 8; ct++) {
        const int k = kbase + ct * 16 + (lane & 15);
        const float w2 = scaled ? v2row[k] : 1.0f;
#pragma unroll
        for (int g = 0; g < 8; g++) {
          const int i = rowLocal + g;
          numacc[c][g] += acc[ct][g] * w2 * Ecd[(size_t)i * NTOK + k];
        }
      }
    }
  }

  // cross-lane reduce within each 16-lane half (rows disjoint per wave/half)
#pragma unroll
  for (int c = 0; c < 5; c++) {
    const int chan = gx * 5 + c;
#pragma unroll
    for (int g = 0; g < 8; g++) {
      float v = numacc[c][g];
      v += __shfl_xor(v, 1);
      v += __shfl_xor(v, 2);
      v += __shfl_xor(v, 4);
      v += __shfl_xor(v, 8);
      if ((lane & 15) == 0) {
        const int i = i0 + wave * 16 + ((lane >> 4) << 3) + g;
        if (chan < DDIM)
          outp[((size_t)b * NTOK + i) * CDIM + h * DDIM + chan] = v;
        else
          denp[(size_t)bh * NTOK + i] = v;
      }
    }
  }
}

// ---------------------------------------------------------------------------
// K4: out /= den (in place)
// ---------------------------------------------------------------------------
__global__ void k_div(float* __restrict__ out, const float* __restrict__ den) {
  const int idx = blockIdx.x * 256 + threadIdx.x;  // 196608 total
  const int bn = idx / CDIM, cc = idx - bn * CDIM;
  const int b = bn / NTOK, n = bn - b * NTOK;
  const int h = cc >> 6;
  out[idx] = out[idx] / den[((size_t)b * HN + h) * NTOK + n];
}

// ---------------------------------------------------------------------------
extern "C" void kernel_launch(void* const* d_in, const int* in_sizes, int n_in,
                              void* d_out, int out_size, void* d_ws, size_t ws_size,
                              hipStream_t stream) {
  const float* x = (const float*)d_in[0];   // [B,N,C] fp32
  const float* W = (const float*)d_in[1];   // [8C,C]  fp32
  float* out = (float*)d_out;               // [B,N,C] fp32

  char* w = (char*)d_ws;                    // ~12.9 MB used
  bf16_t* xbf  = (bf16_t*)(w + 0);          //  393216 B
  bf16_t* Wbf  = (bf16_t*)(w + 393216);     //  655360 B
  bf16_t* abc  = (bf16_t*)(w + 1048576);    // 1179648 B (a,b,c)
  bf16_t* v1t  = (bf16_t*)(w + 2228224);    //  393216 B
  float*  v2t  = (float*) (w + 2621440);    //  786432 B
  bf16_t* eAB  = (bf16_t*)(w + 3407872);    // 2359296 B
  bf16_t* eEFt = (bf16_t*)(w + 5767168);    // 2359296 B
  float*  eCD  = (float*) (w + 8126464);    // 4718592 B
  float*  den  = (float*) (w + 12845056);   //   12288 B

  k_convert<<<2048, 256, 0, stream>>>(x, W, xbf, Wbf);
  k_proj   <<<dim3(10, 6),    256, 0, stream>>>(xbf, Wbf, abc, v1t, v2t);
  k_logits <<<dim3(3, 9, 8),  256, 0, stream>>>(abc, eAB, eCD, eEFt);
  k_triplet<<<dim3(13, 3, 8), 256, 0, stream>>>(eAB, eEFt, eCD, v1t, v2t, out, den);
  k_div    <<<768, 256, 0, stream>>>(out, den);
}
